// FacialLandmarkGNN_72808285602509
// MI455X (gfx1250) — matrix operand order repaired and checked by reading, not compiled, not run on hardware
//
#include <hip/hip_runtime.h>
#include <hip/hip_bf16.h>
#include <math.h>

typedef __attribute__((ext_vector_type(16))) _Float16 v16h;
typedef __attribute__((ext_vector_type(8)))  _Float16 v8h;
typedef __attribute__((ext_vector_type(8)))  float    v8f;
typedef __attribute__((ext_vector_type(4)))  float    v4f;
typedef __attribute__((ext_vector_type(2)))  float    v2f;

#define BB 512
#define NN 478
#define NP 480   // padded K for cached f16 adjacency
#define HH 128
#define BN (BB * NN)

// D = A(16x32 f16) * B(32x16 f16) + C(16x16 f32), wave32
__device__ __forceinline__ v8f wmma_f16(v16h a, v16h b, v8f c) {
  return __builtin_amdgcn_wmma_f32_16x16x32_f16(false, a, false, b, (short)0, c,
                                                false, false);
}

// ---------------------------------------------------------------------------
// messages = A @ h  (per batch).  Block: 64-row tile of one batch element.
// 8 waves: wave w -> rows 16*(w&3).., cols 64*(w>>2)..  (4 WMMA C tiles each).
// MODE 0: load f32 A each time. MODE 1: load f32 A, cache f16 copy.
// MODE 2: load cached f16 A.
// ---------------------------------------------------------------------------
template <int MODE>
__global__ __launch_bounds__(256) void spmm_kernel(
    const float* __restrict__ A, const _Float16* __restrict__ hf,
    _Float16* __restrict__ msg, _Float16* a16) {
  __shared__ _Float16 hT[2][HH * 32];  // h^T K-panel, [col][k], double buffered

  const int b = blockIdx.y;
  const int mTile = blockIdx.x * 64;
  const int tid = threadIdx.x;
  const int lane = tid & 31, wid = tid >> 5;
  const int l15 = lane & 15, hi = lane >> 4;
  const int rG = (wid & 3) * 16, cG = (wid >> 2) * 64;
  const int ar = mTile + rG + l15;  // A row this lane supplies (A-frag layout)
  const bool arOK = ar < NN;

  const float*    Ab   = A   + (size_t)b * NN * NN;
  const _Float16* hb   = hf  + (size_t)b * NN * HH;
  _Float16*       msgb = msg + (size_t)b * NN * HH;
  _Float16*       a16b = a16 + (size_t)b * NN * NP;

  v8f acc[4];
#pragma unroll
  for (int ct = 0; ct < 4; ++ct)
#pragma unroll
    for (int v = 0; v < 8; ++v) acc[ct][v] = 0.f;

  const int KS = (NN + 31) / 32;  // 15 K-steps of 32

  auto stage = [&](int ks) {
    const int k0 = ks * 32;
    int kk = tid >> 4;             // 0..15
    const int c8 = (tid & 15) * 8; // col chunk
    _Float16* dst = &hT[ks & 1][0];
#pragma unroll
    for (int it = 0; it < 2; ++it, kk += 16) {
      const int k = k0 + kk;
      v8h row;
      if (k < NN) {
        row = *(const v8h*)(hb + (size_t)k * HH + c8);  // coalesced 16B
      } else {
#pragma unroll
        for (int j = 0; j < 8; ++j) row[j] = (_Float16)0.f;
      }
#pragma unroll
      for (int j = 0; j < 8; ++j) dst[(c8 + j) * 32 + kk] = row[j];  // transpose
    }
  };

  stage(0);
  for (int ks = 0; ks < KS; ++ks) {
    __syncthreads();
    if (ks + 1 < KS) stage(ks + 1);
    const int k0 = ks * 32;

    // A fragment: lane holds K {kb..kb+15}, kb = hi*16 (contiguous, matches B)
    v16h afrag;
    if (MODE == 2) {
      if (arOK) {
        const _Float16* arow = a16b + (size_t)ar * NP + k0;
        v8h lo = *(const v8h*)(arow + hi * 16);
        v8h hh = *(const v8h*)(arow + 8 + hi * 16);
#pragma unroll
        for (int j = 0; j < 8; ++j) { afrag[j] = lo[j]; afrag[8 + j] = hh[j]; }
      } else {
#pragma unroll
        for (int j = 0; j < 16; ++j) afrag[j] = (_Float16)0.f;
      }
    } else {
      const float* arow = Ab + (size_t)ar * NN;
      if (arOK && (k0 + 32 <= NN)) {
        // Full K-step: unconditional b64 vector loads (rows are 8B aligned).
        const float* p1 = arow + k0 + hi * 16;
        const float* p2 = p1 + 8;
        __builtin_prefetch(p1 + 64, 0, 0);  // stream A ~2 K-steps ahead
#pragma unroll
        for (int j = 0; j < 4; ++j) {
          const v2f a = *(const v2f*)(p1 + 2 * j);
          const v2f c = *(const v2f*)(p2 + 2 * j);
          afrag[2 * j]     = (_Float16)a[0];
          afrag[2 * j + 1] = (_Float16)a[1];
          afrag[8 + 2 * j]     = (_Float16)c[0];
          afrag[8 + 2 * j + 1] = (_Float16)c[1];
        }
      } else {
        // Edge (last K-step or invalid row): per-element guarded loads.
#pragma unroll
        for (int j = 0; j < 8; ++j) {
          const int k1 = k0 + hi * 16 + j;
          const int k2 = k0 + 8 + hi * 16 + j;
          afrag[j]     = (_Float16)((arOK && k1 < NN) ? arow[k1] : 0.f);
          afrag[8 + j] = (_Float16)((arOK && k2 < NN) ? arow[k2] : 0.f);
        }
      }
      if (MODE == 1 && wid < 4 && arOK) {  // waves 0-3 cover all 64 rows once
        v8h lo, hh;
#pragma unroll
        for (int j = 0; j < 8; ++j) { lo[j] = afrag[j]; hh[j] = afrag[8 + j]; }
        *(v8h*)(a16b + (size_t)ar * NP + k0 + hi * 16)      = lo;
        *(v8h*)(a16b + (size_t)ar * NP + k0 + 8 + hi * 16) = hh;
      }
    }

    const _Float16* buf = &hT[ks & 1][0];
#pragma unroll
    for (int ct = 0; ct < 4; ++ct) {
      const int col = cG + ct * 16 + l15;
      // B frag: lane<16 -> K 0..15, lane>=16 -> K 16..31, contiguous in h^T
      v16h bfrag = *(const v16h*)(buf + col * 32 + hi * 16);
      acc[ct] = wmma_f16(afrag, bfrag, acc[ct]);
    }
  }

#pragma unroll
  for (int ct = 0; ct < 4; ++ct) {
    const int col = cG + ct * 16 + l15;
#pragma unroll
    for (int v = 0; v < 8; ++v) {
      const int row = mTile + rG + v + 8 * hi;  // C layout: M = v + 8*hi
      if (row < NN) msgb[(size_t)row * HH + col] = (_Float16)acc[ct][v];
    }
  }
}

// ---------------------------------------------------------------------------
// h_new = relu([h|msg] @ W1 + b1) @ W2 + b2 + h   (64 rows per block)
// ---------------------------------------------------------------------------
__global__ __launch_bounds__(256) void mlp_kernel(
    const _Float16* __restrict__ msg, const float* __restrict__ W1,
    const float* __restrict__ b1, const float* __restrict__ W2,
    const float* __restrict__ b2, float* __restrict__ h32, _Float16* h16) {
  __shared__ _Float16 W1T[HH * 256];  // [col][k]
  __shared__ _Float16 W2T[HH * HH];   // [col][k]
  __shared__ _Float16 u1[64 * HH];    // row-major hidden activations

  const int tid = threadIdx.x;
  const int lane = tid & 31, wid = tid >> 5;
  const int l15 = lane & 15, hi = lane >> 4;
  const int rG = (wid & 3) * 16, cG = (wid >> 2) * 64;
  const size_t mBase = (size_t)blockIdx.x * 64;

  for (int idx = tid; idx < 256 * HH; idx += 256) {
    const int k = idx >> 7, c = idx & 127;
    W1T[c * 256 + k] = (_Float16)W1[idx];
  }
  for (int idx = tid; idx < HH * HH; idx += 256) {
    const int k = idx >> 7, c = idx & 127;
    W2T[c * 128 + k] = (_Float16)W2[idx];
  }
  __syncthreads();

  const size_t rowOff = (mBase + rG + l15) * HH;

  v8f acc[4];
#pragma unroll
  for (int ct = 0; ct < 4; ++ct) {
    const float bv = b1[cG + ct * 16 + l15];
#pragma unroll
    for (int v = 0; v < 8; ++v) acc[ct][v] = bv;
  }

#pragma unroll
  for (int ks = 0; ks < 8; ++ks) {  // K = 256: first 128 from h, next from msg
    const int k0 = ks * 32;
    const _Float16* src = (k0 < HH) ? (const _Float16*)(h16 + rowOff + k0)
                                    : (msg + rowOff + (k0 - HH));
    v8h lo = *(const v8h*)(src + hi * 16);
    v8h hh = *(const v8h*)(src + 8 + hi * 16);
    v16h afrag;
#pragma unroll
    for (int j = 0; j < 8; ++j) { afrag[j] = lo[j]; afrag[8 + j] = hh[j]; }
#pragma unroll
    for (int ct = 0; ct < 4; ++ct) {
      const int col = cG + ct * 16 + l15;
      v16h bfrag = *(const v16h*)(&W1T[col * 256 + k0 + hi * 16]);
      acc[ct] = wmma_f16(afrag, bfrag, acc[ct]);
    }
  }

#pragma unroll
  for (int ct = 0; ct < 4; ++ct) {
    const int col = cG + ct * 16 + l15;
#pragma unroll
    for (int v = 0; v < 8; ++v) {
      float x = acc[ct][v];
      x = x > 0.f ? x : 0.f;  // ReLU
      u1[(rG + v + 8 * hi) * HH + col] = (_Float16)x;
    }
  }
  __syncthreads();

  v8f acc2[4];
#pragma unroll
  for (int ct = 0; ct < 4; ++ct) {
    const float bv = b2[cG + ct * 16 + l15];
#pragma unroll
    for (int v = 0; v < 8; ++v) acc2[ct][v] = bv;
  }

#pragma unroll
  for (int ks = 0; ks < 4; ++ks) {  // K = 128
    const int k0 = ks * 32;
    const _Float16* src = &u1[(rG + l15) * HH + k0];
    v8h lo = *(const v8h*)(src + hi * 16);
    v8h hh = *(const v8h*)(src + 8 + hi * 16);
    v16h afrag;
#pragma unroll
    for (int j = 0; j < 8; ++j) { afrag[j] = lo[j]; afrag[8 + j] = hh[j]; }
#pragma unroll
    for (int ct = 0; ct < 4; ++ct) {
      const int col = cG + ct * 16 + l15;
      v16h bfrag = *(const v16h*)(&W2T[col * 128 + k0 + hi * 16]);
      acc2[ct] = wmma_f16(afrag, bfrag, acc2[ct]);
    }
  }

#pragma unroll
  for (int ct = 0; ct < 4; ++ct) {
    const int col = cG + ct * 16 + l15;
#pragma unroll
    for (int v = 0; v < 8; ++v) {
      const size_t gi = (mBase + rG + v + 8 * hi) * HH + col;
      const float r = acc2[ct][v] + h32[gi];  // residual in f32
      h32[gi] = r;
      h16[gi] = (_Float16)r;
    }
  }
}

// ---------------------------------------------------------------------------
// h0 = relu(x @ enc_w1 + b1) @ enc_w2 + b2   (K of first linear is only 3)
// ---------------------------------------------------------------------------
__global__ __launch_bounds__(256) void encode_kernel(
    const float* __restrict__ x, const float* __restrict__ w1,
    const float* __restrict__ b1, const float* __restrict__ w2,
    const float* __restrict__ b2, float* __restrict__ h32,
    _Float16* __restrict__ h16) {
  __shared__ _Float16 W2T[HH * HH];
  __shared__ _Float16 tls[64 * HH];

  const int tid = threadIdx.x;
  const int lane = tid & 31, wid = tid >> 5;
  const int l15 = lane & 15, hi = lane >> 4;
  const int rG = (wid & 3) * 16, cG = (wid >> 2) * 64;
  const size_t mBase = (size_t)blockIdx.x * 64;

  for (int idx = tid; idx < HH * HH; idx += 256) {
    const int k = idx >> 7, c = idx & 127;
    W2T[c * 128 + k] = (_Float16)w2[idx];
  }
  for (int idx = tid; idx < 64 * HH; idx += 256) {
    const int r = idx >> 7, c = idx & 127;
    const size_t m = mBase + r;
    float t = x[m * 3] * w1[c] + x[m * 3 + 1] * w1[128 + c] +
              x[m * 3 + 2] * w1[256 + c] + b1[c];
    tls[idx] = (_Float16)(t > 0.f ? t : 0.f);
  }
  __syncthreads();

  v8f acc[4];
#pragma unroll
  for (int ct = 0; ct < 4; ++ct) {
    const float bv = b2[cG + ct * 16 + l15];
#pragma unroll
    for (int v = 0; v < 8; ++v) acc[ct][v] = bv;
  }

#pragma unroll
  for (int ks = 0; ks < 4; ++ks) {
    const int k0 = ks * 32;
    const _Float16* src = &tls[(rG + l15) * HH + k0];
    v8h lo = *(const v8h*)(src + hi * 16);
    v8h hh = *(const v8h*)(src + 8 + hi * 16);
    v16h afrag;
#pragma unroll
    for (int j = 0; j < 8; ++j) { afrag[j] = lo[j]; afrag[8 + j] = hh[j]; }
#pragma unroll
    for (int ct = 0; ct < 4; ++ct) {
      const int col = cG + ct * 16 + l15;
      v16h bfrag = *(const v16h*)(&W2T[col * 128 + k0 + hi * 16]);
      acc[ct] = wmma_f16(afrag, bfrag, acc[ct]);
    }
  }

#pragma unroll
  for (int ct = 0; ct < 4; ++ct) {
    const int col = cG + ct * 16 + l15;
#pragma unroll
    for (int v = 0; v < 8; ++v) {
      const size_t gi = (mBase + rG + v + 8 * hi) * HH + col;
      const float r = acc[ct][v];
      h32[gi] = r;
      h16[gi] = (_Float16)r;
    }
  }
}

// ---------------------------------------------------------------------------
// coords = h @ cw + cb ; conf = sigmoid(h @ qw + qb)
// ---------------------------------------------------------------------------
__global__ __launch_bounds__(256) void heads_kernel(
    const float* __restrict__ h32, const float* __restrict__ cw,
    const float* __restrict__ cb, const float* __restrict__ qw,
    const float* __restrict__ qb, float* __restrict__ out) {
  const size_t m = (size_t)blockIdx.x * 256 + threadIdx.x;
  const float* hr = h32 + m * HH;
  float c0 = cb[0], c1 = cb[1], q = qb[0];
#pragma unroll 4
  for (int j = 0; j < HH; j += 4) {
    const v4f hv = *(const v4f*)(hr + j);  // 16B aligned (m*512)
    const v4f qv = *(const v4f*)(qw + j);
#pragma unroll
    for (int jj = 0; jj < 4; ++jj) {
      const v2f cv = *(const v2f*)(cw + 2 * (j + jj));  // contiguous (row of [128,2])
      c0 += hv[jj] * cv[0];
      c1 += hv[jj] * cv[1];
      q += hv[jj] * qv[jj];
    }
  }
  out[2 * m] = c0;
  out[2 * m + 1] = c1;
  out[(size_t)2 * BN + m] = 1.f / (1.f + expf(-q));
}

extern "C" void kernel_launch(void* const* d_in, const int* in_sizes, int n_in,
                              void* d_out, int out_size, void* d_ws,
                              size_t ws_size, hipStream_t stream) {
  (void)in_sizes; (void)n_in; (void)out_size;
  const float* x   = (const float*)d_in[0];
  const float* A   = (const float*)d_in[1];
  const float* ew1 = (const float*)d_in[2];
  const float* eb1 = (const float*)d_in[3];
  const float* ew2 = (const float*)d_in[4];
  const float* eb2 = (const float*)d_in[5];
  const float* gw1 = (const float*)d_in[6];
  const float* gb1 = (const float*)d_in[7];
  const float* gw2 = (const float*)d_in[8];
  const float* gb2 = (const float*)d_in[9];
  const float* cw  = (const float*)d_in[10];
  const float* cb  = (const float*)d_in[11];
  const float* qw  = (const float*)d_in[12];
  const float* qb  = (const float*)d_in[13];

  char* ws = (char*)d_ws;
  const size_t szH32 = (size_t)BN * HH * 4;
  const size_t szH16 = (size_t)BN * HH * 2;
  const size_t szA16 = (size_t)BB * NN * NP * 2;
  float*    h32 = (float*)ws;
  _Float16* h16 = (_Float16*)(ws + szH32);
  _Float16* msg = (_Float16*)(ws + szH32 + szH16);
  const size_t baseNeed = szH32 + 2 * szH16;
  const bool useA16 = ws_size >= baseNeed + szA16;
  _Float16* a16 = (_Float16*)(ws + baseNeed);

  encode_kernel<<<BN / 64, 256, 0, stream>>>(x, ew1, eb1, ew2, eb2, h32, h16);

  const dim3 sg((NN + 63) / 64, BB);
  for (int i = 0; i < 3; ++i) {
    if (i == 0) {
      if (useA16) spmm_kernel<1><<<sg, 256, 0, stream>>>(A, h16, msg, a16);
      else        spmm_kernel<0><<<sg, 256, 0, stream>>>(A, h16, msg, a16);
    } else {
      if (useA16) spmm_kernel<2><<<sg, 256, 0, stream>>>(A, h16, msg, a16);
      else        spmm_kernel<0><<<sg, 256, 0, stream>>>(A, h16, msg, a16);
    }
    mlp_kernel<<<BN / 64, 256, 0, stream>>>(
        msg, gw1 + (size_t)i * 256 * HH, gb1 + i * HH,
        gw2 + (size_t)i * HH * HH, gb2 + i * HH, h32, h16);
  }

  heads_kernel<<<BN / 256, 256, 0, stream>>>(h32, cw, cb, qw, qb, (float*)d_out);
}
